// HAConvGNN_223338299941
// MI455X (gfx1250) — matrix operand order repaired and checked by reading, not compiled, not run on hardware
//
#include <hip/hip_runtime.h>
#include <hip/hip_bf16.h>

typedef __attribute__((ext_vector_type(16))) __bf16 v16bf;
typedef __attribute__((ext_vector_type(2)))  __bf16 v2bf;
typedef __attribute__((ext_vector_type(8)))  float  v8f;

__device__ __forceinline__ __bf16 f2bf(float f) { return static_cast<__bf16>(f); }
__device__ __forceinline__ v2bf f2bf2(float a, float b) {
  v2bf r; r[0] = f2bf(a); r[1] = f2bf(b); return r;
}
__device__ __forceinline__ float rcpf(float x) { return __builtin_amdgcn_rcpf(x); }
__device__ __forceinline__ float fast_sigm(float x) { return rcpf(1.0f + __expf(-x)); }
__device__ __forceinline__ float fast_tanh(float x) { return 1.0f - 2.0f * rcpf(1.0f + __expf(2.0f * x)); }

// ---------------------------------------------------------------------------
// Weight pre-conversion: fp32 [N,K] row-major  ->  bf16 [K, ldp] (transposed,
// N padded to ldp, multiple of 16 so B fragments are single 32B loads).
// ---------------------------------------------------------------------------
__global__ void wconvT_kernel(const float* __restrict__ W, __bf16* __restrict__ out,
                              int N, int K, int ldp) {
  int i = blockIdx.x * 256 + threadIdx.x;
  if (i >= K * ldp) return;
  int k = i / ldp, n = i % ldp;
  out[i] = (n < N) ? f2bf(W[(long)n * K + k]) : f2bf(0.0f);
}
// fp32 [K,N] -> bf16 [K, ldp] (no transpose, pad only)
__global__ void wconvP_kernel(const float* __restrict__ W, __bf16* __restrict__ out,
                              int N, int K, int ldp) {
  int i = blockIdx.x * 256 + threadIdx.x;
  if (i >= K * ldp) return;
  int k = i / ldp, n = i % ldp;
  out[i] = (n < N) ? f2bf(W[(long)k * N + n]) : f2bf(0.0f);
}

// ---------------------------------------------------------------------------
// Tiled GEMM: C = A[M,K]*B (+bias)(+relu), bf16 WMMA / fp32 acc.
// NT consecutive 16x16 N-tiles per wave share one A fragment.
// BMODE: 0 = B fp32 [K,N]; 1 = B fp32 [N,K] (transposed); 2 = B bf16 [K,ldp].
// BIASM: bias by M.  TC: store transposed.
// grid.z batching: Aoff=(z/aDiv)*aStride, Boff=z*bStride (fp32 modes), Coff=z*cStride.
// ---------------------------------------------------------------------------
template <int BMODE, bool RELU, bool BIASM, bool TC, int NT = 2>
__global__ __launch_bounds__(256) void gemm_wmma_kernel(
    const float* __restrict__ A, const void* __restrict__ Bv,
    const float* __restrict__ bias, float* __restrict__ C,
    int M, int N, int K, int lda, int ldb, int ldc,
    long aStride, int aDiv, long bStride, long cStride) {
  int z = blockIdx.z;
  const float* Ab = A + (long)(z / aDiv) * aStride;
  const float* Bf = (const float*)Bv + (BMODE != 2 ? (long)z * bStride : 0);
  const __bf16* Bh = (const __bf16*)Bv;
  float* Cb = C + (long)z * cStride;

  int wave = threadIdx.x >> 5, lane = threadIdx.x & 31;
  int half = lane >> 4, l16 = lane & 15;
  int tilesN = (N + 15) >> 4, tilesM = (M + 15) >> 4;
  int tilesNg = (tilesN + NT - 1) / NT;
  int tile = blockIdx.x * 8 + wave;
  if (tile >= tilesM * tilesNg) return;              // wave-uniform
  int mt = tile / tilesNg, ntb = (tile % tilesNg) * NT;
  int am = mt * 16 + l16;
  bool amOK = am < M;
  const float* Arow = Ab + (long)am * lda;

  auto loadB = [&](int nt, int kb) -> v16bf {
    v16bf r = {};
    int nn = nt * 16 + l16;
#pragma unroll
    for (int e = 0; e < 16; ++e) {
      int kk = kb + (e & 7) + half * 8 + (e >> 3) * 16;
      if (BMODE == 2) { if (kk < K) r[e] = Bh[(long)kk * ldb + nn]; }
      else if (BMODE == 1) r[e] = f2bf((nn < N && kk < K) ? Bf[(long)nn * ldb + kk] : 0.0f);
      else r[e] = f2bf((nn < N && kk < K) ? Bf[(long)kk * ldb + nn] : 0.0f);
    }
    return r;
  };

  v8f acc[NT];
#pragma unroll
  for (int g = 0; g < NT; ++g) acc[g] = v8f{};

  for (int k0 = 0; k0 < K; k0 += 32) {
    if (amOK && k0 + 64 < K) __builtin_prefetch(Arow + k0 + 64, 0, 1);
    v16bf af;
#pragma unroll
    for (int v = 0; v < 8; ++v) {                    // A 16x32 frag, K pairs contiguous
      int kk = (v < 4 ? v * 2 : 16 + (v - 4) * 2) + half * 8 + k0;
      float a0 = 0.0f, a1 = 0.0f;
      if (amOK && kk + 1 < K) { float2 t = *(const float2*)(Arow + kk); a0 = t.x; a1 = t.y; }
      else if (amOK && kk < K) { a0 = Arow[kk]; }
      v2bf pk = f2bf2(a0, a1);
      af[2 * v] = pk[0]; af[2 * v + 1] = pk[1];
    }
    int kb = k0;
#pragma unroll
    for (int g = 0; g < NT; ++g) {
      if (ntb + g < tilesN) {                        // wave-uniform
        v16bf bfr = loadB(ntb + g, kb);
        acc[g] = __builtin_amdgcn_wmma_f32_16x16x32_bf16(false, af, false, bfr,
                                                         (short)0, acc[g], false, false);
      }
    }
  }
#pragma unroll
  for (int g = 0; g < NT; ++g) {
    if (ntb + g >= tilesN) break;
    int nt = ntb + g;
#pragma unroll
    for (int r = 0; r < 8; ++r) {
      int mm = mt * 16 + r + half * 8, nn = nt * 16 + l16;
      if (mm < M && nn < N) {
        float v = acc[g][r];
        if (bias) v += BIASM ? bias[mm] : bias[nn];
        if (RELU) v = fmaxf(v, 0.0f);
        if (TC) Cb[(long)nn * ldc + mm] = v;
        else    Cb[(long)mm * ldc + nn] = v;
      }
    }
  }
}

// ---------------------------------------------------------------------------
// Persistent GRU scan (xg precomputed). 1024 thr = 32 waves. h fp32 + hbf bf16
// in LDS; whhT pre-converted bf16 [256,768]. Wave w: mt=w>>4, 3 N-tiles.
// ---------------------------------------------------------------------------
__global__ __launch_bounds__(1024) void gru_scan_kernel(
    const float* __restrict__ xg, long xgN, long xgT,
    const __bf16* __restrict__ whhT, const float* __restrict__ bhh,
    const float* __restrict__ h0, float* __restrict__ ys, long ysN, long ysT,
    float* __restrict__ hT, int Nb, int T) {
  extern __shared__ float smem[];
  float* h = smem;                         // 32*256 f32
  float* acch = smem + 8192;               // 32*768 f32
  __bf16* hbf = (__bf16*)(smem + 8192 + 24576);  // 32*256 bf16
  int tid = threadIdx.x;
  for (int i = tid; i < 32 * 256; i += 1024) {
    int b = i >> 8;
    float v = (h0 && b < Nb) ? h0[i] : 0.0f;
    h[i] = v; hbf[i] = f2bf(v);
  }
  __syncthreads();
  int wave = tid >> 5, lane = tid & 31, half = lane >> 4, l16 = lane & 15;
  int mt = wave >> 4, ntb = wave & 15;
  const __bf16* hrow = hbf + (mt * 16 + l16) * 256;

  for (int t = 0; t < T; ++t) {
    v8f acc[3] = {v8f{}, v8f{}, v8f{}};
    for (int k0 = 0; k0 < 256; k0 += 32) {
      v16bf af;
#pragma unroll
      for (int v = 0; v < 8; ++v) {
        int kk = (v < 4 ? v * 2 : 16 + (v - 4) * 2) + half * 8 + k0;
        v2bf pk = __builtin_bit_cast(v2bf, *(const unsigned*)(hrow + kk));
        af[2 * v] = pk[0]; af[2 * v + 1] = pk[1];
      }
#pragma unroll
      for (int j = 0; j < 3; ++j) {
        v16bf bfr;
#pragma unroll
        for (int e = 0; e < 16; ++e) {
          int kk = k0 + (e & 7) + half * 8 + (e >> 3) * 16;
          bfr[e] = whhT[(long)kk * 768 + (ntb + 16 * j) * 16 + l16];
        }
        acc[j] = __builtin_amdgcn_wmma_f32_16x16x32_bf16(false, af, false, bfr,
                                                         (short)0, acc[j], false, false);
      }
    }
#pragma unroll
    for (int j = 0; j < 3; ++j)
#pragma unroll
      for (int r = 0; r < 8; ++r) {
        int mm = mt * 16 + r + half * 8, nn = (ntb + 16 * j) * 16 + l16;
        acch[mm * 768 + nn] = acc[j][r] + bhh[nn];
      }
    __syncthreads();
    for (int i = tid; i < Nb * 256; i += 1024) {
      int b = i >> 8, j = i & 255;
      const float* xr = xg + (long)b * xgN + (long)t * xgT;
      float r = fast_sigm(xr[j] + acch[b * 768 + j]);
      float zz = fast_sigm(xr[256 + j] + acch[b * 768 + 256 + j]);
      float n = fast_tanh(xr[512 + j] + r * acch[b * 768 + 512 + j]);
      float hn = (1.0f - zz) * n + zz * h[b * 256 + j];
      h[b * 256 + j] = hn; hbf[b * 256 + j] = f2bf(hn);
      ys[(long)b * ysN + (long)t * ysT + j] = hn;
    }
    __syncthreads();
  }
  if (hT)
    for (int i = tid; i < Nb * 256; i += 1024) hT[i] = h[i];
}

// ---------------------------------------------------------------------------
// gru5..8 (grid.x = s): per-step xg (K=300, A from global fp32 x, B = bf16
// wihT) plus hg (K=256, A from LDS hbf, B = bf16 whhT).
// x layout [b,s,n,e]; outs layout [b,s,t,256]; h0 = tstate.
// ---------------------------------------------------------------------------
__global__ __launch_bounds__(1024) void gru58_scan_kernel(
    const float* __restrict__ x, const __bf16* __restrict__ wihT,
    const __bf16* __restrict__ whhT, const float* __restrict__ bih,
    const float* __restrict__ bhh, const float* __restrict__ h0,
    float* __restrict__ outs) {
  extern __shared__ float smem[];
  float* h = smem;                      // 32*256 f32
  float* accx = smem + 8192;            // 32*768 f32
  float* acch = accx + 24576;           // 32*768 f32
  __bf16* hbf = (__bf16*)(smem + 57344);  // 32*256 bf16
  int s = blockIdx.x, tid = threadIdx.x;
  const __bf16* wihS = wihT + (long)s * 300 * 768;
  const __bf16* whhS = whhT + (long)s * 256 * 768;
  const float* bihS = bih + s * 768;
  const float* bhhS = bhh + s * 768;
  for (int i = tid; i < 32 * 256; i += 1024) { h[i] = h0[i]; hbf[i] = f2bf(h0[i]); }
  __syncthreads();
  int wave = tid >> 5, lane = tid & 31, half = lane >> 4, l16 = lane & 15;
  int mt = wave >> 4, ntb = wave & 15;
  int bb = mt * 16 + l16;                         // batch row (<32)
  const __bf16* hrow = hbf + bb * 256;
  const float* xbase = x + (long)(bb * 4 + s) * 300 * 300;

  for (int t = 0; t < 300; ++t) {
    const float* xrow = xbase + (long)t * 300;
    // ---- xg = x_t @ wih^T (K=300) ----
    v8f ax[3] = {v8f{}, v8f{}, v8f{}};
    for (int k0 = 0; k0 < 300; k0 += 32) {
      v16bf af;
#pragma unroll
      for (int v = 0; v < 8; ++v) {
        int kk = (v < 4 ? v * 2 : 16 + (v - 4) * 2) + half * 8 + k0;
        float a0 = 0.0f, a1 = 0.0f;
        if (kk < 300) { float2 tv = *(const float2*)(xrow + kk); a0 = tv.x; a1 = tv.y; }
        v2bf pk = f2bf2(a0, a1);
        af[2 * v] = pk[0]; af[2 * v + 1] = pk[1];
      }
#pragma unroll
      for (int j = 0; j < 3; ++j) {
        v16bf bfr = {};
#pragma unroll
        for (int e = 0; e < 16; ++e) {
          int kk = k0 + (e & 7) + half * 8 + (e >> 3) * 16;
          if (kk < 300) bfr[e] = wihS[(long)kk * 768 + (ntb + 16 * j) * 16 + l16];
        }
        ax[j] = __builtin_amdgcn_wmma_f32_16x16x32_bf16(false, af, false, bfr,
                                                        (short)0, ax[j], false, false);
      }
    }
    // ---- hg = h @ whh^T (K=256) ----
    v8f ah[3] = {v8f{}, v8f{}, v8f{}};
    for (int k0 = 0; k0 < 256; k0 += 32) {
      v16bf af;
#pragma unroll
      for (int v = 0; v < 8; ++v) {
        int kk = (v < 4 ? v * 2 : 16 + (v - 4) * 2) + half * 8 + k0;
        v2bf pk = __builtin_bit_cast(v2bf, *(const unsigned*)(hrow + kk));
        af[2 * v] = pk[0]; af[2 * v + 1] = pk[1];
      }
#pragma unroll
      for (int j = 0; j < 3; ++j) {
        v16bf bfr;
#pragma unroll
        for (int e = 0; e < 16; ++e) {
          int kk = k0 + (e & 7) + half * 8 + (e >> 3) * 16;
          bfr[e] = whhS[(long)kk * 768 + (ntb + 16 * j) * 16 + l16];
        }
        ah[j] = __builtin_amdgcn_wmma_f32_16x16x32_bf16(false, af, false, bfr,
                                                        (short)0, ah[j], false, false);
      }
    }
#pragma unroll
    for (int j = 0; j < 3; ++j)
#pragma unroll
      for (int r = 0; r < 8; ++r) {
        int mm = mt * 16 + r + half * 8, nn = (ntb + 16 * j) * 16 + l16;
        accx[mm * 768 + nn] = ax[j][r] + bihS[nn];
        acch[mm * 768 + nn] = ah[j][r] + bhhS[nn];
      }
    __syncthreads();
    for (int i = tid; i < 32 * 256; i += 1024) {
      int b = i >> 8, j = i & 255;
      float r = fast_sigm(accx[b * 768 + j] + acch[b * 768 + j]);
      float zz = fast_sigm(accx[b * 768 + 256 + j] + acch[b * 768 + 256 + j]);
      float n = fast_tanh(accx[b * 768 + 512 + j] + r * acch[b * 768 + 512 + j]);
      float hn = (1.0f - zz) * n + zz * h[b * 256 + j];
      h[b * 256 + j] = hn; hbf[b * 256 + j] = f2bf(hn);
      outs[(long)b * 307200 + (long)s * 76800 + (long)t * 256 + j] = hn;
    }
    __syncthreads();
  }
}

// --------------------------- glue kernels ----------------------------------
__global__ void embed_kernel(const int* __restrict__ idx,
                             const float* __restrict__ emb,
                             float* __restrict__ out, int n) {
  int i = blockIdx.x * 256 + threadIdx.x;
  if (i < n) out[i] = emb[(long)idx[i / 300] * 300 + (i % 300)];
}

__global__ void attn_kernel(const float* __restrict__ Q,
                            const float* __restrict__ Kmat, int L,
                            float* __restrict__ ctx, int ctxStride, int ctxOff) {
  __shared__ float S[30 * 50];
  int b = blockIdx.x, tid = threadIdx.x;
  const float* Qb = Q + (long)b * 30 * 256;
  const float* Kb = Kmat + (long)b * L * 256;
  for (int e = tid; e < 30 * L; e += 256) {
    int q = e / L, l = e % L;
    float d = 0.0f;
    for (int k = 0; k < 256; ++k) d += Qb[q * 256 + k] * Kb[l * 256 + k];
    S[e] = d * 0.0625f;
  }
  __syncthreads();
  if (tid < 30) {
    float m = -1e30f;
    for (int l = 0; l < L; ++l) m = fmaxf(m, S[tid * L + l]);
    float sum = 0.0f;
    for (int l = 0; l < L; ++l) { float e = __expf(S[tid * L + l] - m); S[tid * L + l] = e; sum += e; }
    float inv = rcpf(sum);
    for (int l = 0; l < L; ++l) S[tid * L + l] *= inv;
  }
  __syncthreads();
  for (int o = tid; o < 30 * 256; o += 256) {
    int q = o >> 8, d = o & 255;
    float a = 0.0f;
    for (int l = 0; l < L; ++l) a += S[q * L + l] * Kb[l * 256 + d];
    ctx[((long)b * 30 + q) * ctxStride + ctxOff + d] = a;
  }
}

__global__ void ahier_kernel(const float* __restrict__ decout,
                             const float* __restrict__ g4out,
                             float* __restrict__ ahier) {
  __shared__ float S[120];
  int b = blockIdx.x, tid = threadIdx.x;
  for (int e = tid; e < 120; e += 128) {
    int q = e >> 2, s = e & 3;
    const float* qv = decout + ((long)b * 30 + q) * 256;
    const float* av = g4out + (long)b * 1024 + s * 256;
    float d = 0.0f;
    for (int k = 0; k < 256; ++k) d += qv[k] * av[k];
    S[e] = d * 0.0625f;
  }
  __syncthreads();
  if (tid < 30) {
    float m = fmaxf(fmaxf(S[tid * 4], S[tid * 4 + 1]), fmaxf(S[tid * 4 + 2], S[tid * 4 + 3]));
    float e0 = __expf(S[tid * 4 + 0] - m), e1 = __expf(S[tid * 4 + 1] - m);
    float e2 = __expf(S[tid * 4 + 2] - m), e3 = __expf(S[tid * 4 + 3] - m);
    float inv = rcpf(e0 + e1 + e2 + e3);
    float* o = ahier + ((long)b * 30 + tid) * 4;
    o[0] = e0 * inv; o[1] = e1 * inv; o[2] = e2 * inv; o[3] = e3 * inv;
  }
}

__global__ void sc_softmax_kernel(float* __restrict__ sc, const float* __restrict__ ahier) {
  __shared__ float red[128];
  int row = blockIdx.x, tid = threadIdx.x;
  int z = row / 30, q = row % 30, b = z >> 2, s = z & 3;
  float* p = sc + (long)row * 300;
  float mx = -1e30f;
  for (int i = tid; i < 300; i += 128) mx = fmaxf(mx, p[i] * 0.0625f);
  red[tid] = mx; __syncthreads();
  for (int st = 64; st; st >>= 1) { if (tid < st) red[tid] = fmaxf(red[tid], red[tid + st]); __syncthreads(); }
  float M = red[0]; __syncthreads();
  float sum = 0.0f;
  for (int i = tid; i < 300; i += 128) { float e = __expf(p[i] * 0.0625f - M); p[i] = e; sum += e; }
  red[tid] = sum; __syncthreads();
  for (int st = 64; st; st >>= 1) { if (tid < st) red[tid] += red[tid + st]; __syncthreads(); }
  float w = ahier[((long)b * 30 + q) * 4 + s] * rcpf(red[0]);
  for (int i = tid; i < 300; i += 128) p[i] *= w;
}

__global__ void reduce4_kernel(const float* __restrict__ tmpc, float* __restrict__ acp) {
  int i = blockIdx.x * 256 + threadIdx.x;
  if (i >= 245760) return;
  int b = i / 7680, r = i % 7680;
  float v = 0.0f;
  for (int s = 0; s < 4; ++s) v += tmpc[((long)(b * 4 + s)) * 7680 + r];
  acp[i] = v;
}

__global__ void copy_dec_kernel(const float* __restrict__ decout, float* __restrict__ ctx) {
  int i = blockIdx.x * 256 + threadIdx.x;
  if (i >= 245760) return;
  ctx[(long)(i / 256) * 768 + 256 + (i % 256)] = decout[i];
}

// ---------------------------------------------------------------------------
extern "C" void kernel_launch(void* const* d_in, const int* in_sizes, int n_in,
                              void* d_out, int out_size, void* d_ws, size_t ws_size,
                              hipStream_t stream) {
  const int* x0 = (const int*)d_in[0];
  const int* x1 = (const int*)d_in[1];
  const int* x2 = (const int*)d_in[2];
  const float* adj = (const float*)d_in[3];
  const float* emb_ast = (const float*)d_in[4];
  const float* emb_com = (const float*)d_in[5];
  const float *g1_wih = (const float*)d_in[6], *g1_whh = (const float*)d_in[7];
  const float *g1_bih = (const float*)d_in[8], *g1_bhh = (const float*)d_in[9];
  const float *g2_wih = (const float*)d_in[10], *g2_whh = (const float*)d_in[11];
  const float *g2_bih = (const float*)d_in[12], *g2_bhh = (const float*)d_in[13];
  const float *g4_wih = (const float*)d_in[14], *g4_whh = (const float*)d_in[15];
  const float *g4_bih = (const float*)d_in[16], *g4_bhh = (const float*)d_in[17];
  const float *g58_wih = (const float*)d_in[18], *g58_whh = (const float*)d_in[19];
  const float *g58_bih = (const float*)d_in[20], *g58_bhh = (const float*)d_in[21];
  const float *gcn_w = (const float*)d_in[22], *gcn_b = (const float*)d_in[23];
  const float *dense_w = (const float*)d_in[24], *dense_b = (const float*)d_in[25];
  const float *lin_w = (const float*)d_in[26], *lin_b = (const float*)d_in[27];
  float* out = (float*)d_out;

  float* p = (float*)d_ws;
  auto carve = [&](size_t n) { float* r = p; p += n; return r; };
  float* bufA = carve(11520000);   // astwork [b,s,300,300]
  float* bufB = carve(11520000);
  float* bufT = carve(11520000);   // GCN temp, later reused as outs [b,s,300,256]
  float* embx0 = carve(480000);
  float* embx1 = carve(288000);
  float* xg1 = carve(1228800);
  float* xg2 = carve(737280);
  float* xg4 = carve(98304);
  float* tenc = carve(409600);
  float* tstate = carve(8192);
  float* decout = carve(245760);
  float* g4out = carve(32768);     // [b,s,256]
  float* ahier = carve(3840);
  float* scb = carve(1152000);     // [(b*4+s),30,300]
  float* tmpc = carve(983040);     // [(b*4+s),30,256]
  float* acp = carve(245760);
  float* context = carve(737280);  // [b,q,768]
  float* act = carve(245760);
  // bf16 weight staging (transposed to [K, ldp])
  __bf16* g1_wihT = (__bf16*)carve(115200);   // [300,768]
  __bf16* g2_wihT = (__bf16*)carve(115200);   // [300,768]
  __bf16* g1_whhT = (__bf16*)carve(98304);    // [256,768]
  __bf16* g2_whhT = (__bf16*)carve(98304);    // [256,768]
  __bf16* g4_whhT = (__bf16*)carve(98304);    // [256,768]
  __bf16* g58_whhT = (__bf16*)carve(393216);  // 4 x [256,768]
  __bf16* g58_wihT = (__bf16*)carve(460800);  // 4 x [300,768]
  __bf16* gcn_wP = (__bf16*)carve(45600);     // [300,304]
  __bf16* dense_wT = (__bf16*)carve(98304);   // [768,256]
  (void)n_in; (void)in_sizes; (void)out_size; (void)ws_size;

  auto gxN = [](int M, int N, int NT) {
    int tM = (M + 15) / 16, tN = (N + 15) / 16, tNg = (tN + NT - 1) / NT;
    return (unsigned)((tM * tNg + 7) / 8);
  };
  auto cgrid = [](int n) { return (unsigned)((n + 255) / 256); };

  // weight conversions
  wconvT_kernel<<<cgrid(300 * 768), 256, 0, stream>>>(g1_wih, g1_wihT, 768, 300, 768);
  wconvT_kernel<<<cgrid(300 * 768), 256, 0, stream>>>(g2_wih, g2_wihT, 768, 300, 768);
  wconvT_kernel<<<cgrid(256 * 768), 256, 0, stream>>>(g1_whh, g1_whhT, 768, 256, 768);
  wconvT_kernel<<<cgrid(256 * 768), 256, 0, stream>>>(g2_whh, g2_whhT, 768, 256, 768);
  wconvT_kernel<<<cgrid(256 * 768), 256, 0, stream>>>(g4_whh, g4_whhT, 768, 256, 768);
  for (int s = 0; s < 4; ++s) {
    wconvT_kernel<<<cgrid(256 * 768), 256, 0, stream>>>(
        g58_whh + (long)s * 768 * 256, g58_whhT + (long)s * 256 * 768, 768, 256, 768);
    wconvT_kernel<<<cgrid(300 * 768), 256, 0, stream>>>(
        g58_wih + (long)s * 768 * 300, g58_wihT + (long)s * 300 * 768, 768, 300, 768);
  }
  wconvP_kernel<<<cgrid(300 * 304), 256, 0, stream>>>(gcn_w, gcn_wP, 300, 300, 304);
  wconvT_kernel<<<cgrid(768 * 256), 256, 0, stream>>>(dense_w, dense_wT, 256, 768, 256);

  // embeddings
  embed_kernel<<<cgrid(480000), 256, 0, stream>>>(x0, emb_ast, embx0, 480000);
  embed_kernel<<<cgrid(288000), 256, 0, stream>>>(x1, emb_com, embx1, 288000);
  embed_kernel<<<cgrid(11520000), 256, 0, stream>>>(x2, emb_ast, bufA, 11520000);

  // xg precompute: x @ wih^T + bih   (B = bf16 [300,768])
  gemm_wmma_kernel<2, false, false, false><<<dim3(gxN(1600, 768, 2), 1, 1), 256, 0, stream>>>(
      embx0, g1_wihT, g1_bih, xg1, 1600, 768, 300, 300, 768, 768, 0, 1, 0, 0);
  gemm_wmma_kernel<2, false, false, false><<<dim3(gxN(960, 768, 2), 1, 1), 256, 0, stream>>>(
      embx1, g2_wihT, g2_bih, xg2, 960, 768, 300, 300, 768, 768, 0, 1, 0, 0);

  // GCN hop 1: support = astwork @ gcn_w (B bf16 [300,304]); astwork = relu(adj @ support + b)
  gemm_wmma_kernel<2, false, false, false, 4><<<dim3(gxN(38400, 300, 4), 1, 1), 256, 0, stream>>>(
      bufA, gcn_wP, nullptr, bufT, 38400, 300, 300, 300, 304, 300, 0, 1, 0, 0);
  gemm_wmma_kernel<0, true, false, false, 4><<<dim3(gxN(300, 300, 4), 1, 128), 256, 0, stream>>>(
      adj, bufT, gcn_b, bufB, 300, 300, 300, 300, 300, 300, 90000, 1, 90000, 90000);
  // GCN hop 2
  gemm_wmma_kernel<2, false, false, false, 4><<<dim3(gxN(38400, 300, 4), 1, 1), 256, 0, stream>>>(
      bufB, gcn_wP, nullptr, bufT, 38400, 300, 300, 300, 304, 300, 0, 1, 0, 0);
  gemm_wmma_kernel<0, true, false, false, 4><<<dim3(gxN(300, 300, 4), 1, 128), 256, 0, stream>>>(
      adj, bufT, gcn_b, bufA, 300, 300, 300, 300, 300, 300, 90000, 1, 90000, 90000);

  // gru1 (T=50) -> tenc, tstate ; gru2 (T=30, h0=tstate) -> decout
  size_t sh1 = (size_t)(8192 + 24576) * 4 + 8192 * 2;
  gru_scan_kernel<<<1, 1024, sh1, stream>>>(xg1, 38400, 768, g1_whhT, g1_bhh, nullptr,
                                            tenc, 12800, 256, tstate, 32, 50);
  gru_scan_kernel<<<1, 1024, sh1, stream>>>(xg2, 23040, 768, g2_whhT, g2_bhh, tstate,
                                            decout, 7680, 256, nullptr, 32, 30);

  // tcontext -> context[:,:,0:256] ; decout -> context[:,:,256:512]
  attn_kernel<<<32, 256, 0, stream>>>(decout, tenc, 50, context, 768, 0);
  copy_dec_kernel<<<cgrid(245760), 256, 0, stream>>>(decout, context);

  // gru4 xg, computed transposed so the 276MB gru4_wih streams exactly once:
  // xg4^T[768,128] = g4_wih[768,90000] @ astwork^T ; TC store -> xg4[(b*4+s),768]
  gemm_wmma_kernel<1, false, true, true, 8><<<dim3(gxN(768, 128, 8), 1, 1), 256, 0, stream>>>(
      g4_wih, bufA, g4_bih, xg4, 768, 128, 90000, 90000, 90000, 768, 0, 1, 0, 0);
  gru_scan_kernel<<<1, 1024, sh1, stream>>>(xg4, 768, 3072, g4_whhT, g4_bhh, nullptr,
                                            g4out, 256, 1024, nullptr, 4, 32);

  ahier_kernel<<<32, 128, 0, stream>>>(decout, g4out, ahier);

  // gru5..8 over nodes (T=300): fused per-step xg + hg WMMA; outs into bufT
  float* outs = bufT;
  size_t sh58 = (size_t)(8192 + 2 * 24576) * 4 + 8192 * 2;
  gru58_scan_kernel<<<4, 1024, sh58, stream>>>(bufA, g58_wihT, g58_whhT, g58_bih, g58_bhh,
                                               tstate, outs);

  // sc[z=(b*4+s)] = decout[b] @ outs[z]^T ; softmax(scale) * ahier
  gemm_wmma_kernel<1, false, false, false><<<dim3(gxN(30, 300, 2), 1, 128), 256, 0, stream>>>(
      decout, outs, nullptr, scb, 30, 300, 256, 256, 256, 300, 7680, 4, 76800, 9000);
  sc_softmax_kernel<<<3840, 128, 0, stream>>>(scb, ahier);
  // tmpc[z] = atten[z] @ outs[z] ; reduce over s -> acp
  gemm_wmma_kernel<0, false, false, false><<<dim3(gxN(30, 256, 2), 1, 128), 256, 0, stream>>>(
      scb, outs, nullptr, tmpc, 30, 256, 300, 300, 256, 256, 9000, 1, 76800, 7680);
  reduce4_kernel<<<cgrid(245760), 256, 0, stream>>>(tmpc, acp);
  // acontext = dot_attn(decout, acp) -> context[:,:,512:768]
  attn_kernel<<<32, 256, 0, stream>>>(decout, acp, 30, context, 768, 512);

  // dense: relu(context @ dense_w^T + b) -> act (B bf16 [768,256])
  gemm_wmma_kernel<2, true, false, false><<<dim3(gxN(960, 256, 2), 1, 1), 256, 0, stream>>>(
      context, dense_wT, dense_b, act, 960, 256, 768, 768, 256, 256, 0, 1, 0, 0);

  // final: out^T = lin_w @ act^T (+lin_b per row), stored transposed -> out[32,10000]
  gemm_wmma_kernel<1, false, true, true><<<dim3(gxN(10000, 32, 2), 1, 1), 256, 0, stream>>>(
      lin_w, act, lin_b, out, 10000, 32, 7680, 7680, 7680, 10000, 0, 1, 0, 0);
}
